// LSTM_86792699117978
// MI455X (gfx1250) — matrix operand order repaired and checked by reading, not compiled, not run on hardware
//
#include <hip/hip_runtime.h>
#include <hip/hip_bf16.h>

// ---------------------------------------------------------------------------
// 2-layer LSTM for MI455X (gfx1250, wave32, WMMA bf16).
// B=64, S=512, I=H=1024.
//   * bf16 v_wmma_f32_16x16x32_bf16 everywhere, f32 accumulation.
//   * input projection: 128x64-tile WMMA GEMM (8 wmma / 12 b128 per k-step).
//   * recurrent scan: persistent 32 WGs, Wh^T slice (256 KB) LDS-resident for
//     all 512 steps (async-to-LDS staged), double-buffered h, device-wide
//     atomic barrier per step.
// ---------------------------------------------------------------------------

typedef __bf16 bf16_t;
typedef __bf16 v16bf __attribute__((ext_vector_type(16)));
typedef __bf16 v8bf  __attribute__((ext_vector_type(8)));
typedef float  v8f   __attribute__((ext_vector_type(8)));

#define LB   64      // batch
#define LS   512     // seq len
#define LK   1024    // I == H
#define LN   4096    // 4H
#define NWG  32      // persistent workgroups in scan
#define WGC  32      // h columns owned per scan workgroup
#define SCAN_THREADS 256

#if __has_builtin(__builtin_amdgcn_global_load_async_to_lds_b128) && \
    __has_builtin(__builtin_amdgcn_s_wait_asynccnt)
#define HAVE_ASYNC_LDS 1
// Builtin signature (learned from the round-2 diagnostic): param0 is
// AS(1) pointer-to-16B-int-vector, param1 the AS(3) LDS counterpart,
// then immediate offset and cpol.
typedef int v4i_b128 __attribute__((vector_size(16)));
typedef __attribute__((address_space(1))) v4i_b128* g_b128_p;
typedef __attribute__((address_space(3))) v4i_b128* l_b128_p;
#else
#define HAVE_ASYNC_LDS 0
#endif

__device__ __forceinline__ float sigmoidf_(float x) {
    return 1.0f / (1.0f + __expf(-x));
}

// A fragment: 16x32 bf16 tile, row-major source, row = lane&15.
// k order used here (contiguous, same as B): lanes 0-15 hold K 0-15; lanes 16-31 hold K 16-31.
__device__ __forceinline__ v16bf load_a_frag(const bf16_t* __restrict__ base,
                                             int ld, int lane) {
    const bf16_t* p = base + (size_t)(lane & 15) * ld + ((lane >> 4) << 4);
    return *(const v16bf*)(p);
}

// B fragment: 32(K)x16(N) tile read from N-major (transposed) weights WT[n][k].
// k order used here (contiguous, same as A): column n = lane&15; lanes 0-15 hold K 0-15, lanes 16-31 K 16-31.
__device__ __forceinline__ v16bf load_b_frag(const bf16_t* baseT, int ld, int lane) {
    const bf16_t* p = baseT + (size_t)(lane & 15) * ld + ((lane >> 4) << 4);
    return *(const v16bf*)(p);
}

// --------------------------- one-time data prep ----------------------------

// x: [B][S][I] f32  ->  xbf: [S*B][I] bf16 (time-major rows for the GEMM)
__global__ void convert_x_timemajor(const float* __restrict__ x,
                                    bf16_t* __restrict__ xbf) {
    size_t idx = (size_t)blockIdx.x * blockDim.x + threadIdx.x;
    if (idx >= (size_t)LS * LB * LK) return;
    int    i = (int)(idx & (LK - 1));
    size_t r = idx >> 10;            // r = s*64 + b
    int    b = (int)(r & (LB - 1));
    int    s = (int)(r >> 6);
    xbf[idx] = (bf16_t)x[((size_t)b * LS + s) * LK + i];
}

// W: [K][N] f32  ->  WT: [N][K] bf16
__global__ void transpose_to_bf16(const float* __restrict__ W,
                                  bf16_t* __restrict__ WT, int K, int N) {
    size_t idx = (size_t)blockIdx.x * blockDim.x + threadIdx.x;
    if (idx >= (size_t)K * N) return;
    int k = (int)(idx % K);
    int n = (int)(idx / K);
    WT[idx] = (bf16_t)W[(size_t)k * N + n];
}

// ------------------------- input projection GEMM ---------------------------
// C[M][4096] = A[M][1024](bf16) * W(bf16, given as WT[4096][1024]) + bias
// Block tile 128(M) x 64(N), 4 waves; each wave: 2 m-tiles x 4 n-tiles.
__global__ void __launch_bounds__(128)
gemm_xproj(const bf16_t* __restrict__ A, const bf16_t* __restrict__ BT,
           const float* __restrict__ bias, float* __restrict__ C) {
    const int lane = threadIdx.x & 31;
    const int wave = threadIdx.x >> 5;               // 0..3
    const int m0 = blockIdx.y * 128 + wave * 16;     // second m-tile at +64
    const int n0 = blockIdx.x * 64;

    v8f acc[2][4] = {};
    for (int k0 = 0; k0 < LK; k0 += 32) {
        __builtin_prefetch(A + (size_t)m0 * LK + k0 + 64, 0, 1);
        v16bf a0 = load_a_frag(A + (size_t)m0 * LK + k0, LK, lane);
        v16bf a1 = load_a_frag(A + (size_t)(m0 + 64) * LK + k0, LK, lane);
#pragma unroll
        for (int t = 0; t < 4; ++t) {
            v16bf b = load_b_frag(BT + (size_t)(n0 + t * 16) * LK + k0, LK, lane);
            acc[0][t] = __builtin_amdgcn_wmma_f32_16x16x32_bf16(
                false, a0, false, b, (short)0, acc[0][t], false, false);
            acc[1][t] = __builtin_amdgcn_wmma_f32_16x16x32_bf16(
                false, a1, false, b, (short)0, acc[1][t], false, false);
        }
    }
    // C/D layout: m = r + 8*(lane>>4), n = lane&15
    const int mm = (lane >> 4) << 3;
    const int nn = lane & 15;
#pragma unroll
    for (int h = 0; h < 2; ++h) {
#pragma unroll
        for (int t = 0; t < 4; ++t) {
            int   ng = n0 + t * 16 + nn;
            float bv = bias[ng];
#pragma unroll
            for (int r = 0; r < 8; ++r) {
                C[(size_t)(m0 + h * 64 + mm + r) * LN + ng] = acc[h][t][r] + bv;
            }
        }
    }
}

// ----------------------------- recurrent scan ------------------------------
// Persistent: grid = NWG(32) blocks x 256 threads. Block owns h cols
// [col0, col0+32). Wh^T gate strips (4 x 32 x 1024 bf16 = 256 KB) live in LDS.
__global__ void __launch_bounds__(SCAN_THREADS)
lstm_scan(const float* __restrict__ xg,      // [S*B][4096] f32 preacts
          const bf16_t* __restrict__ WhT,    // [4096][1024] bf16
          bf16_t* __restrict__ hbuf,         // [2][64][1024] bf16, zeroed
          bf16_t* __restrict__ hseq,         // [S*B][1024] bf16 or nullptr
          float* __restrict__ outBSH,        // [B][S][H] f32 or nullptr
          unsigned int* __restrict__ bar,    // zeroed counter
          int S) {
    extern __shared__ __align__(32) char smem[];
    bf16_t* sWh   = (bf16_t*)smem;                                 // 256 KB
    float*  sGate = (float*)(smem + 128 * 1024 * sizeof(bf16_t));  //  32 KB
    float*  sC    = sGate + 4 * LB * WGC;                          //   8 KB

    const int tid  = threadIdx.x;
    const int lane = tid & 31;
    const int wave = tid >> 5;               // 0..7
    const int col0 = blockIdx.x * WGC;

    // Load Wh^T slice into LDS once: LDS row r (0..127) = gate (r>>5), col (r&31)
    {
        const int vecPerRow = LK / 8;        // 128 x 16B chunks per row
        for (int idx = tid; idx < 128 * vecPerRow; idx += SCAN_THREADS) {
            int r = idx / vecPerRow;
            int v = idx - r * vecPerRow;
            int n = (r >> 5) * LK + col0 + (r & 31);
            const bf16_t* gp = WhT + (size_t)n * LK + v * 8;
            bf16_t*       lp = sWh + (size_t)r * LK + v * 8;
#if HAVE_ASYNC_LDS
            __builtin_amdgcn_global_load_async_to_lds_b128(
                (g_b128_p)gp, (l_b128_p)lp, 0, 0);
#else
            *(v8bf*)lp = *(const v8bf*)gp;
#endif
        }
#if HAVE_ASYNC_LDS
        __builtin_amdgcn_s_wait_asynccnt(0);
#endif
    }
    for (int i = tid; i < LB * WGC; i += SCAN_THREADS) sC[i] = 0.f;
    __syncthreads();

    const int m0 = (wave & 3) * 16;          // 4 M-blocks of 16
    const int nh = wave >> 2;                // 2 halves of the 128 gate cols

    for (int s = 0; s < S; ++s) {
        const bf16_t* hprev = hbuf + (size_t)(s & 1) * LB * LK;
        bf16_t*       hnext = hbuf + (size_t)((s & 1) ^ 1) * LB * LK;
        const float*  xg_s  = xg + (size_t)s * LB * LN;

        // warm L1/L2 for this step's 32KB xg tile (one line per thread)
        __builtin_prefetch(xg_s + (size_t)(tid >> 2) * LN + (tid & 3) * 1024 + col0,
                           0, 0);

        v8f acc[4] = {};
        for (int k0 = 0; k0 < LK; k0 += 32) {
            v16bf a = load_a_frag(hprev + (size_t)m0 * LK + k0, LK, lane);
#pragma unroll
            for (int t = 0; t < 4; ++t) {
                int nt = nh * 4 + t;         // n-tile 0..7 over 128 LDS cols
                v16bf b = load_b_frag(sWh + (size_t)(nt * 16) * LK + k0, LK, lane);
                acc[t] = __builtin_amdgcn_wmma_f32_16x16x32_bf16(
                    false, a, false, b, (short)0, acc[t], false, false);
            }
        }
        // stage h@Wh pre-activations in LDS
        const int mm = (lane >> 4) << 3;
        const int nn = lane & 15;
#pragma unroll
        for (int t = 0; t < 4; ++t) {
            int nl = (nh * 4 + t) * 16 + nn; // 0..127
            int g  = nl >> 5;
            int c  = nl & 31;
#pragma unroll
            for (int r = 0; r < 8; ++r)
                sGate[((g * LB) + m0 + mm + r) * WGC + c] = acc[t][r];
        }
        __syncthreads();

        // fused gates + cell/state update (2048 cells / 256 threads)
#pragma unroll
        for (int it = 0; it < (LB * WGC) / SCAN_THREADS; ++it) {
            int cell = tid + it * SCAN_THREADS;
            int m = cell >> 5;
            int c = cell & 31;
            const float* xr = xg_s + (size_t)m * LN + col0 + c;
            float pi = sGate[(0 * LB + m) * WGC + c] + xr[0];
            float pf = sGate[(1 * LB + m) * WGC + c] + xr[1024];
            float pg = sGate[(2 * LB + m) * WGC + c] + xr[2048];
            float po = sGate[(3 * LB + m) * WGC + c] + xr[3072];
            float i_ = sigmoidf_(pi);
            float f_ = sigmoidf_(pf);
            float g_ = tanhf(pg);
            float o_ = sigmoidf_(po);
            float cn = f_ * sC[cell] + i_ * g_;
            sC[cell] = cn;
            float h = o_ * tanhf(cn);
            hnext[(size_t)m * LK + col0 + c] = (bf16_t)h;
            if (hseq)   hseq[((size_t)(s * LB + m)) * LK + col0 + c] = (bf16_t)h;
            if (outBSH) outBSH[((size_t)m * S + s) * LK + col0 + c] = h;
        }

        // device-wide barrier: all 32 WGs must publish h before next step
        __threadfence();
        __syncthreads();
        if (tid == 0) {
            atomicAdd(bar, 1u);
            unsigned target = (unsigned)(s + 1) * NWG;
            while (__hip_atomic_load(bar, __ATOMIC_ACQUIRE,
                                     __HIP_MEMORY_SCOPE_AGENT) < target)
                __builtin_amdgcn_s_sleep(2);
        }
        __syncthreads();
    }
}

// ------------------------------- launcher ----------------------------------
extern "C" void kernel_launch(void* const* d_in, const int* in_sizes, int n_in,
                              void* d_out, int out_size, void* d_ws,
                              size_t ws_size, hipStream_t stream) {
    (void)in_sizes; (void)n_in; (void)out_size; (void)ws_size;
    const float* x   = (const float*)d_in[0];
    const float* Wx0 = (const float*)d_in[1];
    const float* Wh0 = (const float*)d_in[2];
    const float* b0  = (const float*)d_in[3];
    const float* Wx1 = (const float*)d_in[4];
    const float* Wh1 = (const float*)d_in[5];
    const float* b1  = (const float*)d_in[6];
    float* out = (float*)d_out;

    char*  ws  = (char*)d_ws;
    size_t off = 0;
    auto carve = [&](size_t bytes) -> char* {
        char* p = ws + off;
        off += (bytes + 255) & ~(size_t)255;
        return p;
    };
    bf16_t* xbf  = (bf16_t*)carve((size_t)LS * LB * LK * 2);   // 64 MB
    bf16_t* hseq = (bf16_t*)carve((size_t)LS * LB * LK * 2);   // 64 MB
    bf16_t* WxT0 = (bf16_t*)carve((size_t)LN * LK * 2);        //  8 MB
    bf16_t* WhT0 = (bf16_t*)carve((size_t)LN * LK * 2);
    bf16_t* WxT1 = (bf16_t*)carve((size_t)LN * LK * 2);
    bf16_t* WhT1 = (bf16_t*)carve((size_t)LN * LK * 2);
    float*  xg   = (float*)carve((size_t)LS * LB * LN * 4);    // 512 MB (reused)
    bf16_t* hbuf = (bf16_t*)carve((size_t)2 * LB * LK * 2);    // 256 KB
    unsigned int* bars = (unsigned int*)carve(256);

    (void)hipMemsetAsync(bars, 0, 8, stream);
    (void)hipMemsetAsync(hbuf, 0, (size_t)2 * LB * LK * 2, stream);

    { // data prep
        size_t nx = (size_t)LS * LB * LK;
        convert_x_timemajor<<<dim3((unsigned)((nx + 255) / 256)), 256, 0, stream>>>(x, xbf);
        size_t nw = (size_t)LN * LK;
        unsigned wb = (unsigned)((nw + 255) / 256);
        transpose_to_bf16<<<wb, 256, 0, stream>>>(Wx0, WxT0, LK, LN);
        transpose_to_bf16<<<wb, 256, 0, stream>>>(Wh0, WhT0, LK, LN);
        transpose_to_bf16<<<wb, 256, 0, stream>>>(Wx1, WxT1, LK, LN);
        transpose_to_bf16<<<wb, 256, 0, stream>>>(Wh1, WhT1, LK, LN);
    }

    dim3 g1(LN / 64, (LS * LB) / 128);  // (64, 256)
    size_t scanShm = (size_t)128 * 1024 * 2 + 4 * LB * WGC * 4 + LB * WGC * 4;

    // Layer 0
    gemm_xproj<<<g1, 128, 0, stream>>>(xbf, WxT0, b0, xg);
    lstm_scan<<<NWG, SCAN_THREADS, scanShm, stream>>>(
        xg, WhT0, hbuf, hseq, nullptr, bars + 0, LS);

    // Layer 1
    (void)hipMemsetAsync(hbuf, 0, (size_t)2 * LB * LK * 2, stream);
    gemm_xproj<<<g1, 128, 0, stream>>>(hseq, WxT1, b1, xg);
    lstm_scan<<<NWG, SCAN_THREADS, scanShm, stream>>>(
        xg, WhT1, hbuf, nullptr, out, bars + 1, LS);
}
